// ParticleDynamicsModel_38955353374984
// MI455X (gfx1250) — matrix operand order repaired and checked by reading, not compiled, not run on hardware
//
#include <hip/hip_runtime.h>

// ---------------------------------------------------------------------------
// Problem constants (from reference): B=4, N=8192, D=16, R=32, H=64, E=262144
// ---------------------------------------------------------------------------
#define PB 4
#define PN 8192
#define PD 16
#define PR 32
#define PH 64
#define PE 262144

typedef __attribute__((ext_vector_type(16))) _Float16 v16h;
typedef __attribute__((ext_vector_type(2)))  _Float16 v2h;
typedef __attribute__((ext_vector_type(2)))  __fp16   v2fp16;
typedef __attribute__((ext_vector_type(8)))  float    v8f;

union V16 { v16h v; v2h p[8]; };

__device__ __forceinline__ v2h pk2(float a, float b) {
  v2fp16 r = __builtin_amdgcn_cvt_pkrtz(a, b);   // v_cvt_pkrtz_f16_f32
  return __builtin_bit_cast(v2h, r);
}

// Exchange a 32-bit value between lane L and lane L^16 (SWAPX16 pattern):
// ds_swizzle_b32 offset = {xor=0x10, or=0, and=0x1f} -> 0x401F
__device__ __forceinline__ v2h swapx16(v2h x) {
  int i = __builtin_bit_cast(int, x);
  i = __builtin_amdgcn_ds_swizzle(i, 0x401f);
  return __builtin_bit_cast(v2h, i);
}

__device__ __forceinline__ v8f wmma_f16(const v16h& a, const v16h& b, const v8f& c) {
  return __builtin_amdgcn_wmma_f32_16x16x32_f16(false, a, false, b, (short)0, c,
                                                false, false);
}

// ---------------------------------------------------------------------------
// Kernel 0: zero the rel accumulator (B*N*R floats = 4 MB) in workspace
// ---------------------------------------------------------------------------
__global__ void zero_kernel(float4* __restrict__ p, int n4) {
  int i = blockIdx.x * blockDim.x + threadIdx.x;
  int stride = gridDim.x * blockDim.x;
  for (; i < n4; i += stride) p[i] = make_float4(0.f, 0.f, 0.f, 0.f);
}

// ---------------------------------------------------------------------------
// Kernel 1: edge MLP + scatter-add.
// Per 16-edge tile (one wave):
//   B-matrix (32x16 f16) = edge_in^T : lanes 0-15 carry sender feats (K=0..15),
//                                      lanes 16-31 carry receiver feats (K=16..31)
//   h^T  = relu(W1^T x^T + b1)   : 4x v_wmma_f32_16x16x32_f16 (M tiles of H=64)
//   rel^T= relu(W2^T h^T + b2)   : 4x WMMA (2 K-chunks x 2 M-tiles), h^T fed
//                                  through ds_swizzle SWAPX16 to become B-layout
//   scatter: global_atomic_add_f32 into rel[b, receivers[e], :]
// ---------------------------------------------------------------------------
__global__ __launch_bounds__(256) void edge_kernel(
    const float* __restrict__ particles, const int* __restrict__ senders,
    const int* __restrict__ receivers, const float* __restrict__ W1,
    const float* __restrict__ b1, const float* __restrict__ W2,
    const float* __restrict__ b2, float* __restrict__ rel) {
  const int lane = threadIdx.x & 31;
  const int hf   = lane >> 4;     // half-wave index
  const int mrow = lane & 15;     // row (A) / column (B,C) within tile
  const int kb   = hf ? 8 : 0;    // A-layout K base for this half

  // ---- preload W1^T as four 16x32 A-tiles (f16, A-layout) ----
  V16 A1[4];
#pragma unroll
  for (int mt = 0; mt < 4; ++mt) {
    const int m = 16 * mt + mrow;
#pragma unroll
    for (int j = 0; j < 4; ++j) {
      A1[mt].p[j]     = pk2(W1[(16 * hf + 2 * j) * PH + m],     W1[(16 * hf + 2 * j + 1) * PH + m]);
      A1[mt].p[4 + j] = pk2(W1[(16 * hf + 8 + 2 * j) * PH + m], W1[(16 * hf + 9 + 2 * j) * PH + m]);
    }
  }
  // ---- preload W2^T as 2 (M) x 2 (K-chunk) A-tiles ----
  V16 A2[2][2];
#pragma unroll
  for (int mt = 0; mt < 2; ++mt) {
    const int m = 16 * mt + mrow;
#pragma unroll
    for (int kc = 0; kc < 2; ++kc) {
      const int k0 = 32 * kc + 16 * hf;
#pragma unroll
      for (int j = 0; j < 4; ++j) {
        A2[mt][kc].p[j]     = pk2(W2[(k0 + 2 * j) * PR + m],      W2[(k0 + 2 * j + 1) * PR + m]);
        A2[mt][kc].p[4 + j] = pk2(W2[(k0 + 8 + 2 * j) * PR + m], W2[(k0 + 9 + 2 * j) * PR + m]);
      }
    }
  }
  // ---- biases in C-layout (VGPR i -> row 16*mt + i + 8*hf) ----
  float b1v[4][8], b2v[2][8];
#pragma unroll
  for (int mt = 0; mt < 4; ++mt)
#pragma unroll
    for (int i = 0; i < 8; ++i) b1v[mt][i] = b1[16 * mt + i + 8 * hf];
#pragma unroll
  for (int mt = 0; mt < 2; ++mt)
#pragma unroll
    for (int i = 0; i < 8; ++i) b2v[mt][i] = b2[16 * mt + i + 8 * hf];

  const v8f z = {0.f, 0.f, 0.f, 0.f, 0.f, 0.f, 0.f, 0.f};
  const int ntiles = PB * (PE / 16);                      // 65536
  const int wid = (blockIdx.x * blockDim.x + threadIdx.x) >> 5;
  const int nw  = (gridDim.x * blockDim.x) >> 5;

  for (int t = wid; t < ntiles; t += nw) {
    const int b = t >> 14;                                // E/16 = 16384 tiles per batch
    const int e = ((t & 16383) << 4) + mrow;
    const int rv = receivers[e];
    const int sv = senders[e];
    const int idx = hf ? rv : sv;

    // gather 16 contiguous features of one particle per lane
    const float4* fp = (const float4*)(particles + (((size_t)b * PN + idx) << 4));
    float4 f0 = fp[0], f1 = fp[1], f2 = fp[2], f3 = fp[3];
    V16 Bv;
    Bv.p[0] = pk2(f0.x, f0.y); Bv.p[1] = pk2(f0.z, f0.w);
    Bv.p[2] = pk2(f1.x, f1.y); Bv.p[3] = pk2(f1.z, f1.w);
    Bv.p[4] = pk2(f2.x, f2.y); Bv.p[5] = pk2(f2.z, f2.w);
    Bv.p[6] = pk2(f3.x, f3.y); Bv.p[7] = pk2(f3.z, f3.w);

    // GEMM1: h^T tiles (H rows 16*mt..16*mt+15, cols = 16 edges)
    v8f h[4];
#pragma unroll
    for (int mt = 0; mt < 4; ++mt) {
      h[mt] = wmma_f16(A1[mt].v, Bv.v, z);
#pragma unroll
      for (int i = 0; i < 8; ++i) h[mt][i] = fmaxf(h[mt][i] + b1v[mt][i], 0.0f);
    }

    // GEMM2: rel^T = W2^T * h^T ; h^T (C-layout) -> B-layout via SWAPX16
    v8f r0 = z, r1 = z;
#pragma unroll
    for (int kc = 0; kc < 2; ++kc) {
      const int tA = 2 * kc, tB = tA + 1;
      v2h pA[4], pB[4], got[4];
#pragma unroll
      for (int j = 0; j < 4; ++j) {
        pA[j] = pk2(h[tA][2 * j], h[tA][2 * j + 1]);
        pB[j] = pk2(h[tB][2 * j], h[tB][2 * j + 1]);
      }
#pragma unroll
      for (int j = 0; j < 4; ++j) got[j] = swapx16(hf ? pA[j] : pB[j]);
      V16 B2;
#pragma unroll
      for (int j = 0; j < 4; ++j) {
        B2.p[j]     = hf ? got[j] : pA[j];
        B2.p[4 + j] = hf ? pB[j]  : got[j];
      }
      r0 = wmma_f16(A2[0][kc].v, B2.v, r0);
      r1 = wmma_f16(A2[1][kc].v, B2.v, r1);
    }

    // bias + relu + atomic scatter into rel[b, rv, :]
    float* rb = rel + (((size_t)b * PN + rv) << 5);
#pragma unroll
    for (int i = 0; i < 8; ++i) {
      float v0 = fmaxf(r0[i] + b2v[0][i], 0.0f);
      float v1 = fmaxf(r1[i] + b2v[1][i], 0.0f);
      atomicAdd(rb + (i + 8 * hf), v0);
      atomicAdd(rb + (16 + i + 8 * hf), v1);
    }
  }
}

// ---------------------------------------------------------------------------
// Kernel 2: node MLP.
//   dyn_in = [particles(16) | rel(32)] (K=48, zero-padded to 64)
//   h2^T   = relu(W3^T dyn^T + b3)  : 8 WMMAs
//   delta^T= W4^T h2^T              : 2 WMMAs (SWAPX16 re-layout again)
//   out    = particles + delta + b4
// ---------------------------------------------------------------------------
__global__ __launch_bounds__(256) void node_kernel(
    const float* __restrict__ particles, const float* __restrict__ W3,
    const float* __restrict__ b3, const float* __restrict__ W4,
    const float* __restrict__ b4, const float* __restrict__ rel,
    float* __restrict__ out) {
  const int lane = threadIdx.x & 31;
  const int hf   = lane >> 4;
  const int mrow = lane & 15;
  const int kb   = hf ? 8 : 0;

  // W3^T: 4 (M) x 2 (K-chunk) A-tiles, zero-padded for k >= 48
  V16 A3[4][2];
#pragma unroll
  for (int mt = 0; mt < 4; ++mt) {
    const int m = 16 * mt + mrow;
#pragma unroll
    for (int kc = 0; kc < 2; ++kc) {
#pragma unroll
      for (int j = 0; j < 4; ++j) {
        const int k0 = 32 * kc + 16 * hf + 2 * j;
        const int k2 = 32 * kc + 16 * hf + 8 + 2 * j;
        float x0 = (k0 < 48)     ? W3[k0 * PH + m]       : 0.f;
        float x1 = (k0 + 1 < 48) ? W3[(k0 + 1) * PH + m] : 0.f;
        float x2 = (k2 < 48)     ? W3[k2 * PH + m]       : 0.f;
        float x3 = (k2 + 1 < 48) ? W3[(k2 + 1) * PH + m] : 0.f;
        A3[mt][kc].p[j]     = pk2(x0, x1);
        A3[mt][kc].p[4 + j] = pk2(x2, x3);
      }
    }
  }
  // W4^T: single M tile (D=16), 2 K-chunks
  V16 A4[2];
#pragma unroll
  for (int kc = 0; kc < 2; ++kc) {
#pragma unroll
    for (int j = 0; j < 4; ++j) {
      const int k0 = 32 * kc + 16 * hf + 2 * j;
      const int k2 = 32 * kc + 16 * hf + 8 + 2 * j;
      A4[kc].p[j]     = pk2(W4[k0 * PD + mrow], W4[(k0 + 1) * PD + mrow]);
      A4[kc].p[4 + j] = pk2(W4[k2 * PD + mrow], W4[(k2 + 1) * PD + mrow]);
    }
  }
  float b3v[4][8], b4v[8];
#pragma unroll
  for (int mt = 0; mt < 4; ++mt)
#pragma unroll
    for (int i = 0; i < 8; ++i) b3v[mt][i] = b3[16 * mt + i + 8 * hf];
#pragma unroll
  for (int i = 0; i < 8; ++i) b4v[i] = b4[i + 8 * hf];

  const v8f z = {0.f, 0.f, 0.f, 0.f, 0.f, 0.f, 0.f, 0.f};
  const v2h zz = {(_Float16)0.f, (_Float16)0.f};
  const int ntiles = PB * (PN / 16);                      // 2048
  const int wid = (blockIdx.x * blockDim.x + threadIdx.x) >> 5;
  const int nw  = (gridDim.x * blockDim.x) >> 5;

  for (int t = wid; t < ntiles; t += nw) {
    const int b = t >> 9;                                 // N/16 = 512 tiles per batch
    const int n = ((t & 511) << 4) + mrow;
    const float* pbase = particles + (((size_t)b * PN + n) << 4);
    const float* rbase = rel + (((size_t)b * PN + n) << 5);

    // K-chunk 0: lanes 0-15 particles[0..15], lanes 16-31 rel[0..15]
    const float4* fsrc = (const float4*)(hf ? rbase : pbase);
    float4 f0 = fsrc[0], f1 = fsrc[1], f2 = fsrc[2], f3 = fsrc[3];
    // K-chunk 1: lanes 0-15 rel[16..31], lanes 16-31 zero pad
    const float4* gsrc = (const float4*)(rbase + 16);
    float4 g0 = gsrc[0], g1 = gsrc[1], g2 = gsrc[2], g3 = gsrc[3];

    V16 B0, B1;
    B0.p[0] = pk2(f0.x, f0.y); B0.p[1] = pk2(f0.z, f0.w);
    B0.p[2] = pk2(f1.x, f1.y); B0.p[3] = pk2(f1.z, f1.w);
    B0.p[4] = pk2(f2.x, f2.y); B0.p[5] = pk2(f2.z, f2.w);
    B0.p[6] = pk2(f3.x, f3.y); B0.p[7] = pk2(f3.z, f3.w);
    B1.p[0] = hf ? zz : pk2(g0.x, g0.y); B1.p[1] = hf ? zz : pk2(g0.z, g0.w);
    B1.p[2] = hf ? zz : pk2(g1.x, g1.y); B1.p[3] = hf ? zz : pk2(g1.z, g1.w);
    B1.p[4] = hf ? zz : pk2(g2.x, g2.y); B1.p[5] = hf ? zz : pk2(g2.z, g2.w);
    B1.p[6] = hf ? zz : pk2(g3.x, g3.y); B1.p[7] = hf ? zz : pk2(g3.z, g3.w);

    v8f h2[4];
#pragma unroll
    for (int mt = 0; mt < 4; ++mt) {
      v8f acc = wmma_f16(A3[mt][0].v, B0.v, z);
      acc = wmma_f16(A3[mt][1].v, B1.v, acc);
#pragma unroll
      for (int i = 0; i < 8; ++i) acc[i] = fmaxf(acc[i] + b3v[mt][i], 0.0f);
      h2[mt] = acc;
    }

    v8f d = z;
#pragma unroll
    for (int kc = 0; kc < 2; ++kc) {
      const int tA = 2 * kc, tB = tA + 1;
      v2h pA[4], pB[4], got[4];
#pragma unroll
      for (int j = 0; j < 4; ++j) {
        pA[j] = pk2(h2[tA][2 * j], h2[tA][2 * j + 1]);
        pB[j] = pk2(h2[tB][2 * j], h2[tB][2 * j + 1]);
      }
#pragma unroll
      for (int j = 0; j < 4; ++j) got[j] = swapx16(hf ? pA[j] : pB[j]);
      V16 Bh;
#pragma unroll
      for (int j = 0; j < 4; ++j) {
        Bh.p[j]     = hf ? got[j] : pA[j];
        Bh.p[4 + j] = hf ? pB[j]  : got[j];
      }
      d = wmma_f16(A4[kc].v, Bh.v, d);
    }

    // out = particles + delta + b4 ; each lane writes 8 contiguous floats
    float4 pa = *(const float4*)(pbase + 8 * hf);
    float4 pc = *(const float4*)(pbase + 8 * hf + 4);
    float* ob = out + (((size_t)b * PN + n) << 4) + 8 * hf;
    *(float4*)ob = make_float4(d[0] + b4v[0] + pa.x, d[1] + b4v[1] + pa.y,
                               d[2] + b4v[2] + pa.z, d[3] + b4v[3] + pa.w);
    *(float4*)(ob + 4) = make_float4(d[4] + b4v[4] + pc.x, d[5] + b4v[5] + pc.y,
                                     d[6] + b4v[6] + pc.z, d[7] + b4v[7] + pc.w);
  }
}

// ---------------------------------------------------------------------------
// Launch
// ---------------------------------------------------------------------------
extern "C" void kernel_launch(void* const* d_in, const int* in_sizes, int n_in,
                              void* d_out, int out_size, void* d_ws, size_t ws_size,
                              hipStream_t stream) {
  const float* particles = (const float*)d_in[0];
  const int*   senders   = (const int*)d_in[1];
  const int*   receivers = (const int*)d_in[2];
  const float* W1 = (const float*)d_in[3];
  const float* b1 = (const float*)d_in[4];
  const float* W2 = (const float*)d_in[5];
  const float* b2 = (const float*)d_in[6];
  const float* W3 = (const float*)d_in[7];
  const float* b3 = (const float*)d_in[8];
  const float* W4 = (const float*)d_in[9];
  const float* b4 = (const float*)d_in[10];
  float* out = (float*)d_out;
  float* rel = (float*)d_ws;   // B*N*R floats = 4 MB scratch accumulator

  const int rel_f4 = (PB * PN * PR) / 4;
  zero_kernel<<<512, 256, 0, stream>>>((float4*)rel, rel_f4);
  edge_kernel<<<512, 256, 0, stream>>>(particles, senders, receivers,
                                       W1, b1, W2, b2, rel);
  node_kernel<<<256, 256, 0, stream>>>(particles, W3, b3, W4, b4, rel, out);
}
